// ContrastiveLoss_42125039239359
// MI455X (gfx1250) — matrix operand order repaired and checked by reading, not compiled, not run on hardware
//
#include <hip/hip_runtime.h>
#include <hip/hip_bf16.h>
#include <hip/hip_fp16.h>

typedef __attribute__((ext_vector_type(16))) _Float16 v16h;
typedef __attribute__((ext_vector_type(8)))  _Float16 v8h;
typedef __attribute__((ext_vector_type(8)))  float    v8f;
typedef __attribute__((ext_vector_type(4)))  unsigned int u32x4;
typedef __attribute__((ext_vector_type(8)))  int      i32x8;
typedef __attribute__((ext_vector_type(4)))  int      i32x4;

#define DIM       256
#define KTILES    (DIM / 32)             // 8 WMMA k-steps of 32
#define EXP_SCALE 20.6099290241325967f   // (1/0.07) * log2(e): fold T into v_exp_f32
#define LN2F      0.6931471805599453f

// ---------------------------------------------------------------------------
// Kernel 1: L2-normalize rows of [nrows, 256] fp32 -> f16 matrix in workspace.
// ---------------------------------------------------------------------------
__global__ __launch_bounds__(256) void normalize_rows_f16(
    const float* __restrict__ emb, _Float16* __restrict__ out, int nrows) {
  const int wave = threadIdx.x >> 5;
  const int lane = threadIdx.x & 31;
  const int row  = blockIdx.x * 8 + wave;
  if (row >= nrows) return;

  const float4* p = (const float4*)(emb + (size_t)row * DIM) + lane * 2;
  float4 x0 = p[0];
  float4 x1 = p[1];
  float ss = x0.x * x0.x + x0.y * x0.y + x0.z * x0.z + x0.w * x0.w +
             x1.x * x1.x + x1.y * x1.y + x1.z * x1.z + x1.w * x1.w;
#pragma unroll
  for (int off = 16; off >= 1; off >>= 1) ss += __shfl_xor(ss, off, 32);
  const float inv = 1.0f / fmaxf(sqrtf(ss), 1e-12f);

  v8h o;
  o[0] = (_Float16)(x0.x * inv); o[1] = (_Float16)(x0.y * inv);
  o[2] = (_Float16)(x0.z * inv); o[3] = (_Float16)(x0.w * inv);
  o[4] = (_Float16)(x1.x * inv); o[5] = (_Float16)(x1.y * inv);
  o[6] = (_Float16)(x1.z * inv); o[7] = (_Float16)(x1.w * inv);
  *(v8h*)(out + (size_t)row * DIM + lane * 8) = o;
}

// ---------------------------------------------------------------------------
// Fragment loaders for v_wmma_f32_16x16x32_f16 (wave32).
// A (16x32, MxK): lane m = lane&15, half = lane>>4.
//   elems [0..7]  = K (ktile*32 + half*8 .. +7), [8..15] = +16.
// B (32x16, KxN): lane n = lane&15, half = lane>>4.
//   elems [0..7]  = K (ktile*32 + half*8 .. +7), [8..15] = +16.
// ---------------------------------------------------------------------------
__device__ __forceinline__ v16h load_frag_A(const _Float16* __restrict__ rowp,
                                            int ktile, int hf) {
  const _Float16* p = rowp + ktile * 32 + hf * 8;
  v8h lo = *(const v8h*)(p);
  v8h hi = *(const v8h*)(p + 16);
  v16h f;
#pragma unroll
  for (int i = 0; i < 8; ++i) { f[i] = lo[i]; f[i + 8] = hi[i]; }
  return f;
}

__device__ __forceinline__ v16h load_frag_B(const _Float16* __restrict__ rowp,
                                            int ktile, int hf) {
  const _Float16* p = rowp + ktile * 32 + hf * 8;
  v8h lo = *(const v8h*)(p);       // ds_load_b128 (LDS source)
  v8h hi = *(const v8h*)(p + 16);
  v16h f;
#pragma unroll
  for (int i = 0; i < 8; ++i) { f[i] = lo[i]; f[i + 8] = hi[i]; }
  return f;
}

// ---------------------------------------------------------------------------
// TDM: DMA one 32-row x 256-col f16 tile (16 KB) from global into LDS.
// D# per CDNA5 ISA: 2D, data_size=2B, tensor_dim0=256,
// stride=256, tile = 256 x 32. Groups 2/3 zero. Tracked by TENSORcnt.
// ---------------------------------------------------------------------------
__device__ __forceinline__ void tdm_load_tile32(const _Float16* gsrc,
                                                unsigned lds_bytes) {
  unsigned long long ga = (unsigned long long)(__UINTPTR_TYPE__)gsrc;
  u32x4 g0 = { 1u,                                   // count=1 (valid user D#)
               lds_bytes,                            // lds_addr
               (unsigned)ga,                         // global_addr[31:0]
               (unsigned)((ga >> 32) & 0x01FFFFFFu) | (2u << 30) }; // addr[56:32] | type=2
  i32x8 g1 = { (int)(1u << 16),        // workgroup_mask=0, data_size=1 (2 bytes)
               (int)(256u << 16),      // atomic_barrier_addr=0 | tensor_dim0.lo16=256
               (int)(32u << 16),       // tensor_dim0.hi16=0    | tensor_dim1.lo16=32
               (int)(256u << 16),      // tensor_dim1.hi16=0    | tile_dim0=256
               32,                     // tile_dim1=32, tile_dim2=0
               256,                    // tensor_dim0_stride[31:0]=256
               (int)(256u << 16),      // dim0_stride.hi16=0 | tensor_dim1_stride.lo16=256
               0 };                    // tensor_dim1_stride.hi32=0
  i32x4 g2 = { 0, 0, 0, 0 };
  i32x4 g3 = { 0, 0, 0, 0 };
#if defined(__clang_major__) && (__clang_major__ >= 23)
  i32x8 g4 = { 0, 0, 0, 0, 0, 0, 0, 0 };
  __builtin_amdgcn_tensor_load_to_lds(g0, g1, g2, g3, g4, 0);
#else
  __builtin_amdgcn_tensor_load_to_lds(g0, g1, g2, g3, 0);
#endif
}

// ---------------------------------------------------------------------------
// Kernel 2: block = 256 threads (8 waves) owns 64 rows (4 row-tiles of 16).
//   wave = {stream(1b), rowtile(2b)}. Two TDM streams (even/odd 32-col tiles)
//   driven by waves 0 and 4, double-buffered in LDS; 4 waves share each tile.
//   One workgroup barrier per 32 columns; B fragments staged to registers so
//   the 8 WMMAs per 16-col subtile issue back-to-back after one dscnt wait.
// ---------------------------------------------------------------------------
__global__ __launch_bounds__(256) void contrastive_wmma(
    const _Float16* __restrict__ Nmat, const int* __restrict__ ids,
    float* __restrict__ row_loss, float* __restrict__ row_valid, int nrows) {
  __shared__ _Float16 btile[2][2][32 * DIM];   // [stream][buf]: 4 x 16 KB = 64 KB

  const int wave    = threadIdx.x >> 5;
  const int lane    = threadIdx.x & 31;
  const int hf      = lane >> 4;
  const int l16     = lane & 15;
  const int stream  = wave >> 2;                 // 0: even 32-col tiles, 1: odd
  const int rt      = wave & 3;                  // row-tile within block
  const bool leader = (rt == 0);                 // waves 0 and 4 issue TDM
  const int rowtile0 = blockIdx.x * 64 + rt * 16;
  const int ntiles32 = nrows >> 5;               // 256 tiles of 32 columns
  const int iters    = ntiles32 >> 1;            // 128 per stream

  // A fragments (this wave's 16 rows): loaded once, reused for every column.
  v16h a[KTILES];
  {
    const _Float16* rp = Nmat + (size_t)(rowtile0 + l16) * DIM;
#pragma unroll
    for (int t = 0; t < KTILES; ++t) a[t] = load_frag_A(rp, t, hf);
  }

  int idrow[8];
#pragma unroll
  for (int v = 0; v < 8; ++v) idrow[v] = ids[rowtile0 + v + 8 * hf];

  float accAll[8], accPos[8];
#pragma unroll
  for (int v = 0; v < 8; ++v) { accAll[v] = 0.f; accPos[v] = 0.f; }

  // Prime: 32-col tile 'stream' -> buffer 0.
  if (leader)
    tdm_load_tile32(Nmat + (size_t)stream * 32 * DIM,
                    (unsigned)(__UINTPTR_TYPE__)&btile[stream][0][0]);

  int p = 0;
  for (int i = 0; i < iters; ++i) {
    if (leader) __builtin_amdgcn_s_wait_tensorcnt(0); // tile i landed in buf[p]
    __syncthreads();  // publishes buf[p]; also: all waves done with buf[p^1]
    if (leader && (i + 1 < iters)) {                  // overlap tile i+1 DMA
      const int nt = 2 * (i + 1) + stream;
      tdm_load_tile32(Nmat + (size_t)nt * 32 * DIM,
                      (unsigned)(__UINTPTR_TYPE__)&btile[stream][p ^ 1][0]);
    }

    const int tbase = (2 * i + stream) * 32;
#pragma unroll
    for (int sub = 0; sub < 2; ++sub) {
      const _Float16* bt = &btile[stream][p][(size_t)(sub * 16 + l16) * DIM];

      // Stage all 8 B fragments in registers, then run the WMMA chain.
      v16h bfr[KTILES];
#pragma unroll
      for (int t = 0; t < KTILES; ++t) bfr[t] = load_frag_B(bt, t, hf);

      v8f c = {};
#pragma unroll
      for (int t = 0; t < KTILES; ++t)
        c = __builtin_amdgcn_wmma_f32_16x16x32_f16(
            false, a[t], false, bfr[t], (short)0, c, false, false);

      const int ncol = tbase + sub * 16 + l16;
      const int idn  = ids[ncol];
#pragma unroll
      for (int v = 0; v < 8; ++v) {
        const int m = rowtile0 + v + 8 * hf;
        float e = __builtin_amdgcn_exp2f(c[v] * EXP_SCALE); // exp(sim/T)
        e = (m == ncol) ? 0.0f : e;                         // diagonal -> 0
        accAll[v] += e;
        accPos[v] += (idn == idrow[v]) ? e : 0.0f;
      }
    }
    p ^= 1;
  }

  // Reduce across the 16 lanes of each half-group (xor 1..8 stays in-group).
#pragma unroll
  for (int v = 0; v < 8; ++v) {
#pragma unroll
    for (int off = 1; off < 16; off <<= 1) {
      accAll[v] += __shfl_xor(accAll[v], off, 32);
      accPos[v] += __shfl_xor(accPos[v], off, 32);
    }
  }

  // Overlay reduction scratch onto btile (main loop done, no DMA in flight).
  __syncthreads();
  float* red_all = (float*)&btile[0][0][0];   // 128 floats: [wave][16]
  float* red_pos = red_all + 128;             // 128 floats
  if (l16 == 0) {
#pragma unroll
    for (int v = 0; v < 8; ++v) {
      red_all[wave * 16 + v + 8 * hf] = accAll[v];
      red_pos[wave * 16 + v + 8 * hf] = accPos[v];
    }
  }
  __syncthreads();

  // Threads 0..63 combine the two column streams and finish their row.
  if (threadIdx.x < 64) {
    const int lr  = threadIdx.x;
    const int rtt = lr >> 4;
    const int r16 = lr & 15;
    const float sa = red_all[rtt * 16 + r16] + red_all[(rtt + 4) * 16 + r16];
    const float sp = red_pos[rtt * 16 + r16] + red_pos[(rtt + 4) * 16 + r16];
    const bool valid = sp > 0.0f;   // exp > 0, so sp>0 <=> a positive exists
    const float ll = (__builtin_amdgcn_logf(sa) - __builtin_amdgcn_logf(sp)) * LN2F;
    row_loss[blockIdx.x * 64 + lr]  = valid ? ll : 0.0f;
    row_valid[blockIdx.x * 64 + lr] = valid ? 1.0f : 0.0f;
  }
}

// ---------------------------------------------------------------------------
// Kernel 3: deterministic tree reduction of per-row results -> scalar loss.
// ---------------------------------------------------------------------------
__global__ __launch_bounds__(256) void finalize_loss(
    const float* __restrict__ row_loss, const float* __restrict__ row_valid,
    float* __restrict__ out, int nrows) {
  __shared__ float sl[256];
  __shared__ float sv[256];
  float l = 0.f, v = 0.f;
  for (int i = threadIdx.x; i < nrows; i += 256) { l += row_loss[i]; v += row_valid[i]; }
  sl[threadIdx.x] = l; sv[threadIdx.x] = v;
  __syncthreads();
  for (int s = 128; s > 0; s >>= 1) {
    if (threadIdx.x < (unsigned)s) {
      sl[threadIdx.x] += sl[threadIdx.x + s];
      sv[threadIdx.x] += sv[threadIdx.x + s];
    }
    __syncthreads();
  }
  if (threadIdx.x == 0) out[0] = sl[0] / fmaxf(sv[0], 1.0f);
}

// ---------------------------------------------------------------------------
extern "C" void kernel_launch(void* const* d_in, const int* in_sizes, int n_in,
                              void* d_out, int out_size, void* d_ws, size_t ws_size,
                              hipStream_t stream) {
  const float* emb = (const float*)d_in[0];   // [nrows, 256] fp32
  const int*   ids = (const int*)d_in[1];     // [nrows] int32
  float*       out = (float*)d_out;           // scalar

  const int nrows = in_sizes[1];              // 8192

  // Workspace: per-row loss f32 | per-row valid f32 | normalized f16 matrix.
  char* ws = (char*)d_ws;
  float* row_loss  = (float*)ws;
  float* row_valid = (float*)(ws + (size_t)4 * nrows);
  size_t f16_off   = (((size_t)8 * nrows) + 255) & ~(size_t)255;
  _Float16* nmat   = (_Float16*)(ws + f16_off);

  normalize_rows_f16<<<(nrows + 7) / 8, 256, 0, stream>>>(emb, nmat, nrows);
  contrastive_wmma<<<nrows / 64, 256, 0, stream>>>(nmat, ids, row_loss, row_valid, nrows);
  finalize_loss<<<1, 256, 0, stream>>>(row_loss, row_valid, out, nrows);
}
